// Llama2Attention_83889301226120
// MI455X (gfx1250) — matrix operand order repaired and checked by reading, not compiled, not run on hardware
//
#include <hip/hip_runtime.h>
#include <hip/hip_bf16.h>

// ---------------------------------------------------------------------------
// Llama2 attention forward for MI455X (gfx1250, wave32, WMMA).
// Pipeline: fp32->bf16 convert, QKV GEMMs (v_wmma_f32_16x16x32_bf16),
// RoPE, flash-attention (online softmax, WMMA QK^T and PV), output GEMM.
// Data movement: TDM (tensor_load_to_lds) for GEMM A tiles,
// global_load_async_to_lds_b128 for attention K tiles, when available.
// ---------------------------------------------------------------------------

typedef __attribute__((ext_vector_type(16))) __bf16 v16bf;
typedef __attribute__((ext_vector_type(8)))  float  v8f;
typedef __attribute__((ext_vector_type(4)))  unsigned int v4u;
typedef __attribute__((ext_vector_type(8)))  int v8i;
typedef __attribute__((ext_vector_type(4)))  int v4i;

#define B_    2
#define S_    2048
#define D_    4096
#define NH_   32
#define NKV_  8
#define HD_   128
#define NREP_ 4
#define MTOT  (B_ * S_)

#if __has_builtin(__builtin_amdgcn_global_load_async_to_lds_b128)
#define HAVE_ASYNC 1
#else
#define HAVE_ASYNC 0
#endif

#if __has_builtin(__builtin_amdgcn_tensor_load_to_lds)
#define HAVE_TDM 1
#else
#define HAVE_TDM 0
#endif

#if __has_builtin(__builtin_amdgcn_s_wait_asynccnt)
#define WAIT_ASYNC0() __builtin_amdgcn_s_wait_asynccnt(0)
#else
#define WAIT_ASYNC0() asm volatile("s_wait_asynccnt 0" ::: "memory")
#endif

#if __has_builtin(__builtin_amdgcn_s_wait_tensorcnt)
#define WAIT_TENSOR0() __builtin_amdgcn_s_wait_tensorcnt(0)
#else
#define WAIT_TENSOR0() asm volatile("s_wait_tensorcnt 0" ::: "memory")
#endif

__device__ __forceinline__ unsigned short f32_to_bf16(float f) {
  unsigned int u = __float_as_uint(f);
  u += 0x7FFFu + ((u >> 16) & 1u);        // round-to-nearest-even
  return (unsigned short)(u >> 16);
}
__device__ __forceinline__ float bf16_to_f32(unsigned short h) {
  return __uint_as_float(((unsigned int)h) << 16);
}

// Low 32 bits of a flat LDS pointer are the LDS byte offset on AMDGPU.
__device__ __forceinline__ unsigned lds_offset_of(const void* p) {
  return (unsigned)(uintptr_t)p;
}

#if HAVE_ASYNC
// The builtin's prototype (from the round-2 diagnostic) takes typed pointers:
//   arg0: v4i in the global (__device__, AS1) address space
//   arg1: v4i in the LDS (AS3) address space
typedef __attribute__((address_space(1))) v4i g_v4i;
typedef __attribute__((address_space(3))) v4i l_v4i;
__device__ __forceinline__ void async_copy_b128(unsigned short* lds_dst,
                                                const unsigned short* g_src) {
  __builtin_amdgcn_global_load_async_to_lds_b128(
      (g_v4i*)(uintptr_t)g_src,
      (l_v4i*)(uintptr_t)(unsigned)(uintptr_t)lds_dst,
      0, 0);
}
#endif

#if HAVE_TDM
// 2D bf16 tile load via the Tensor Data Mover.
//   tile: tile_k x tile_rows elements, source row stride = row_stride (elems),
//   LDS destination padded by pad_amount_code DWORDs every pad_interval_code
//   (D# encoding: interval 0=2DW,1=4,2=8,3=16...; amount 0=1DW,1=2,...).
__device__ __forceinline__ void tdm_load_tile_2d(unsigned lds_off, const void* gptr,
                                                 int tile_k, int tile_rows,
                                                 int row_stride, int tensor_d0,
                                                 int tensor_d1,
                                                 unsigned pad_interval_code,
                                                 unsigned pad_amount_code,
                                                 int pad_en) {
  unsigned long long ga = (unsigned long long)(uintptr_t)gptr;
  v4u g0 = {1u,                                   // count=1, user descriptor
            lds_off,                              // lds_addr
            (unsigned)ga,                         // global_addr[31:0]
            (unsigned)((ga >> 32) & 0x1FFFFFFu) | (2u << 30)};  // ga[56:32]|type=2
  unsigned w0 = (1u << 16);                       // data_size = 2 bytes
  if (pad_en) w0 |= (1u << 20) | (pad_interval_code << 22) | (pad_amount_code << 25);
  v8i g1 = {
      (int)w0,
      (int)(((unsigned)tensor_d0 & 0xFFFFu) << 16),                 // tensor_dim0 lo
      (int)((((unsigned)tensor_d0 >> 16) & 0xFFFFu) |
            (((unsigned)tensor_d1 & 0xFFFFu) << 16)),               // d0 hi | d1 lo
      (int)((((unsigned)tensor_d1 >> 16) & 0xFFFFu) |
            ((unsigned)tile_k << 16)),                              // d1 hi | tile_dim0
      (int)((unsigned)tile_rows),                                   // tile_dim1 (dim2=0)
      (int)((unsigned)row_stride),                                  // dim0_stride lo32
      0, 0};
  v4i gz = {0, 0, 0, 0};
#if __clang_major__ >= 23
  v8i gz8 = {0, 0, 0, 0, 0, 0, 0, 0};
  __builtin_amdgcn_tensor_load_to_lds(g0, g1, gz, gz, gz8, 0);
#else
  __builtin_amdgcn_tensor_load_to_lds(g0, g1, gz, gz, 0);
#endif
}
#endif

union ABFrag { unsigned int u[8]; v16bf v; };

__device__ __forceinline__ v8f zero_v8f() {
  v8f z = {0.f, 0.f, 0.f, 0.f, 0.f, 0.f, 0.f, 0.f};
  return z;
}

// A fragment: 16(M) x 32(K) bf16. Rows at base + m*stride + koff (row-major).
// ISA layout: lanes 0-15 (M=lane): VGPR0-3 -> K 0..7, VGPR4-7 -> K 16..23;
//             lanes 16-31: K 8..15 and 24..31. Pairs contiguous in K.
__device__ __forceinline__ v16bf load_frag_a(const unsigned short* base, int stride,
                                             int koff, int lane) {
  const int m = lane & 15, kh = lane >> 4;
  const unsigned short* row = base + (size_t)m * stride + koff;
  ABFrag f;
#pragma unroll
  for (int r = 0; r < 8; ++r) {
    int k0 = (r < 4) ? (kh * 8 + 2 * r) : (16 + kh * 8 + 2 * (r - 4));
    f.u[r] = *(const unsigned int*)(row + k0);
  }
  return f.v;
}

// B fragment: 32(K) x 16(N) bf16 gathered from an [n][k] (K-contiguous) layout.
// ISA layout: lanes 0-15 (N=lane): VGPR0-3 -> K 0..7, VGPR4-7 -> K 16..23;
//             lanes 16-31: K 8..15 and 24..31 (same K order as A).
__device__ __forceinline__ v16bf load_frag_b(const unsigned short* base, int stride,
                                             int koff, int lane) {
  const int n = lane & 15, kh = lane >> 4;
  const unsigned short* row = base + (size_t)n * stride + koff;
  ABFrag f;
#pragma unroll
  for (int r = 0; r < 8; ++r) {
    int k0 = (r < 4) ? (kh * 8 + 2 * r) : (16 + kh * 8 + 2 * (r - 4));
    f.u[r] = *(const unsigned int*)(row + k0);
  }
  return f.v;
}

// ---------------------------------------------------------------------------
// fp32 -> bf16 elementwise convert
// ---------------------------------------------------------------------------
__global__ __launch_bounds__(256)
void cvt_f32_bf16(const float* __restrict__ in, unsigned short* __restrict__ out, size_t n) {
  size_t i = (size_t)blockIdx.x * blockDim.x + threadIdx.x;
  size_t stride = (size_t)gridDim.x * blockDim.x;
  for (; i < n; i += stride) out[i] = f32_to_bf16(in[i]);
}

// ---------------------------------------------------------------------------
// bf16 GEMM: C(MxN) = A(MxK,row-major) * B(KxN,row-major), fp32 accumulate.
// 128x128 block tile, BK=32, 256 threads = 8 waves, wave tile 64x32.
// A tile staged by TDM (or async copies); B tile staged transposed manually.
// ---------------------------------------------------------------------------
template <bool OUT_BF16>
__global__ __launch_bounds__(256)
void gemm_bf16_wmma(const unsigned short* __restrict__ A,
                    const unsigned short* __restrict__ Bm,
                    void* __restrict__ Cv, int M, int N, int K) {
  __shared__ __attribute__((aligned(16))) unsigned short As[128 * 40];  // 16DW row + 4DW pad
  __shared__ __attribute__((aligned(16))) unsigned short Bs[128 * 34];  // transposed [n][k]
  const int tid = threadIdx.x, lane = tid & 31, wave = tid >> 5;
  const int wr = wave >> 2, wc = wave & 3;
  const int blockM = blockIdx.y * 128, blockN = blockIdx.x * 128;

  v8f acc[4][2];
#pragma unroll
  for (int i = 0; i < 4; ++i)
#pragma unroll
    for (int j = 0; j < 2; ++j) acc[i][j] = zero_v8f();

#if HAVE_TDM
  const unsigned as_lds = lds_offset_of(&As[0]);
#endif

  for (int kk = 0; kk < K; kk += 32) {
    // ---- Stage A tile (128 rows x 32 bf16), LDS row stride 40 elems ----
#if HAVE_TDM
    if (wave == 0) {
      tdm_load_tile_2d(as_lds, A + (size_t)blockM * K + kk,
                       /*tile_k=*/32, /*tile_rows=*/128,
                       /*row_stride=*/K, /*tensor_d0=*/K, /*tensor_d1=*/M,
                       /*pad_interval(16DW)=*/3u, /*pad_amount(4DW)=*/3u, 1);
    }
#elif HAVE_ASYNC
#pragma unroll
    for (int i = 0; i < 2; ++i) {
      int c = tid + i * 256;
      int row = c >> 2, col = (c & 3) * 8;
      async_copy_b128(As + row * 40 + col, A + (size_t)(blockM + row) * K + kk + col);
    }
#else
#pragma unroll
    for (int i = 0; i < 2; ++i) {
      int c = tid + i * 256;
      int row = c >> 2, col = (c & 3) * 8;
      *(uint4*)(As + row * 40 + col) =
          *(const uint4*)(A + (size_t)(blockM + row) * K + kk + col);
    }
#endif
    // ---- Stage B tile (32x128) transposed into Bs[n][k] ----
#pragma unroll
    for (int i = 0; i < 2; ++i) {
      int c = tid + i * 256;
      int kr = c >> 4, n8 = (c & 15) * 8;
      uint4 vv = *(const uint4*)(Bm + (size_t)(kk + kr) * N + blockN + n8);
      const unsigned short* e = (const unsigned short*)&vv;
#pragma unroll
      for (int j = 0; j < 8; ++j) Bs[(n8 + j) * 34 + kr] = e[j];
    }
#if HAVE_TDM
    if (wave == 0) WAIT_TENSOR0();
#elif HAVE_ASYNC
    WAIT_ASYNC0();
#endif
    __syncthreads();

    if (kk + 32 < K) {  // prefetch next tiles (global_prefetch_b8)
      __builtin_prefetch(A + (size_t)(blockM + (tid >> 1)) * K + kk + 32, 0, 0);
      __builtin_prefetch(Bm + (size_t)(kk + 32 + (tid >> 4)) * N + blockN + (tid & 15) * 8, 0, 0);
    }

    v16bf af[4], bfr[2];
#pragma unroll
    for (int i = 0; i < 4; ++i) af[i] = load_frag_a(As + (wr * 64 + i * 16) * 40, 40, 0, lane);
#pragma unroll
    for (int j = 0; j < 2; ++j) bfr[j] = load_frag_b(Bs + (wc * 32 + j * 16) * 34, 34, 0, lane);
#pragma unroll
    for (int i = 0; i < 4; ++i)
#pragma unroll
      for (int j = 0; j < 2; ++j)
        acc[i][j] = __builtin_amdgcn_wmma_f32_16x16x32_bf16(
            false, af[i], false, bfr[j], (short)0, acc[i][j], false, false);
    __syncthreads();
  }

  // C layout: lane holds column n = lane&15; VGPR v holds row v + 8*(lane>>4).
  const int n = lane & 15, mo = (lane >> 4) * 8;
#pragma unroll
  for (int i = 0; i < 4; ++i)
#pragma unroll
    for (int j = 0; j < 2; ++j) {
      int gm = blockM + wr * 64 + i * 16 + mo;
      int gn = blockN + wc * 32 + j * 16 + n;
#pragma unroll
      for (int vv = 0; vv < 8; ++vv) {
        float val = acc[i][j][vv];
        if (OUT_BF16)
          ((unsigned short*)Cv)[(size_t)(gm + vv) * N + gn] = f32_to_bf16(val);
        else
          ((float*)Cv)[(size_t)(gm + vv) * N + gn] = val;
      }
    }
}

// ---------------------------------------------------------------------------
// RoPE on bf16 tensor (MTOT, nh, HD): rotate adjacent (even,odd) pairs.
// ---------------------------------------------------------------------------
__global__ __launch_bounds__(256)
void rope_bf16(unsigned short* __restrict__ t, const float* __restrict__ cosT,
               const float* __restrict__ sinT, int nh, size_t npairs) {
  size_t idx = (size_t)blockIdx.x * blockDim.x + threadIdx.x;
  if (idx >= npairs) return;
  const int half = HD_ / 2;
  int i = (int)(idx % half);
  size_t rest = idx / half;
  int h = (int)(rest % nh);
  size_t bs = rest / nh;
  int s = (int)(bs % S_);
  unsigned short* p = t + (bs * (size_t)nh + h) * HD_ + 2 * i;
  float a = bf16_to_f32(p[0]), b2 = bf16_to_f32(p[1]);
  float c = cosT[(size_t)s * half + i], sn = sinT[(size_t)s * half + i];
  p[0] = f32_to_bf16(a * c - b2 * sn);
  p[1] = f32_to_bf16(a * sn + b2 * c);
}

// ---------------------------------------------------------------------------
// Flash attention (no causal mask, matching reference). One block per
// (b*NH, qblock of 64). 4 waves, each owns 16 query rows. GQA via kvh = h/NREP.
// K tile staged via async global->LDS copies when available.
// ---------------------------------------------------------------------------
__global__ __launch_bounds__(128)
void attn_wmma(const unsigned short* __restrict__ q, const unsigned short* __restrict__ k,
               const unsigned short* __restrict__ v, unsigned short* __restrict__ o) {
  __shared__ __attribute__((aligned(16))) unsigned short Ks[32 * 136];  // [key][hd]
  __shared__ __attribute__((aligned(16))) unsigned short Vs[128 * 34];  // [hd][key]
  __shared__ __attribute__((aligned(16))) unsigned short Ps[4][16 * 34];
  const int tid = threadIdx.x, lane = tid & 31, wave = tid >> 5;
  const int bh = blockIdx.x;
  const int b = bh / NH_, h = bh % NH_, kvh = h / NREP_;
  const int q0 = blockIdx.y * 64 + wave * 16;
  const float scale = 0.08838834764831845f;  // 1/sqrt(128)
  const int qstride = NH_ * HD_;
  const int kvstride = NKV_ * HD_;

  // Q fragments (16 rows x 128), gathered straight from global; held all loop.
  const unsigned short* qbase = q + ((size_t)(b * S_ + q0)) * qstride + h * HD_;
  v16bf qf[4];
#pragma unroll
  for (int t4 = 0; t4 < 4; ++t4) qf[t4] = load_frag_a(qbase, qstride, t4 * 32, lane);

  v8f Oacc[8];
#pragma unroll
  for (int t = 0; t < 8; ++t) Oacc[t] = zero_v8f();
  float mrow[8], lrow[8];
#pragma unroll
  for (int vv = 0; vv < 8; ++vv) { mrow[vv] = -1e30f; lrow[vv] = 0.f; }

  const unsigned short* kbase = k + (size_t)b * S_ * kvstride + kvh * HD_;
  const unsigned short* vbase = v + (size_t)b * S_ * kvstride + kvh * HD_;

  for (int kb0 = 0; kb0 < S_; kb0 += 32) {
    // Stage K chunk [32 keys][128 hd] (natural layout == QK^T B-operand layout).
#pragma unroll
    for (int i = 0; i < 4; ++i) {
      int c = tid + i * 128;
      int key = c >> 4, h8 = (c & 15) * 8;
#if HAVE_ASYNC
      async_copy_b128(Ks + key * 136 + h8,
                      kbase + (size_t)(kb0 + key) * kvstride + h8);
#else
      *(uint4*)(Ks + key * 136 + h8) =
          *(const uint4*)(kbase + (size_t)(kb0 + key) * kvstride + h8);
#endif
    }
    // Stage V chunk transposed -> Vs[hd][key] (PV B-operand layout).
#pragma unroll
    for (int i = 0; i < 4; ++i) {
      int c = tid + i * 128;
      int key = c >> 4, h8 = (c & 15) * 8;
      uint4 val = *(const uint4*)(vbase + (size_t)(kb0 + key) * kvstride + h8);
      const unsigned short* e = (const unsigned short*)&val;
#pragma unroll
      for (int j = 0; j < 8; ++j) Vs[(h8 + j) * 34 + key] = e[j];
    }
#if HAVE_ASYNC
    WAIT_ASYNC0();
#endif
    __syncthreads();

    // S = Q * K^T for 2 key tiles of 16; contraction over HD via 4 WMMA each.
    v8f st[2];
    st[0] = zero_v8f(); st[1] = zero_v8f();
#pragma unroll
    for (int nt = 0; nt < 2; ++nt)
#pragma unroll
      for (int t4 = 0; t4 < 4; ++t4) {
        v16bf bfr = load_frag_b(Ks + (nt * 16) * 136, 136, t4 * 32, lane);
        st[nt] = __builtin_amdgcn_wmma_f32_16x16x32_bf16(
            false, qf[t4], false, bfr, (short)0, st[nt], false, false);
      }

    // Online softmax. Each C-tile row lives in one 16-lane half, so xor-shuffles
    // with masks 8..1 reduce across exactly the lanes holding that row.
    float corr[8];
#pragma unroll
    for (int vv = 0; vv < 8; ++vv) {
      float x = fmaxf(st[0][vv], st[1][vv]) * scale;
#pragma unroll
      for (int off2 = 8; off2 >= 1; off2 >>= 1) x = fmaxf(x, __shfl_xor(x, off2, 32));
      float mnew = fmaxf(mrow[vv], x);
      float p0 = __expf(st[0][vv] * scale - mnew);
      float p1 = __expf(st[1][vv] * scale - mnew);
      st[0][vv] = p0; st[1][vv] = p1;
      float rsum = p0 + p1;
#pragma unroll
      for (int off2 = 8; off2 >= 1; off2 >>= 1) rsum += __shfl_xor(rsum, off2, 32);
      float cr = __expf(mrow[vv] - mnew);
      lrow[vv] = lrow[vv] * cr + rsum;
      mrow[vv] = mnew;
      corr[vv] = cr;
    }
#pragma unroll
    for (int t = 0; t < 8; ++t)
#pragma unroll
      for (int vv = 0; vv < 8; ++vv) Oacc[t][vv] *= corr[vv];

    // P (C layout, fp32) -> per-wave LDS -> A layout bf16. LDS ops are in-order
    // within a wave, and the buffer is private to this wave: no barrier needed.
    {
      const int n = lane & 15, kh2 = lane >> 4;
      unsigned short* pw = Ps[wave];
#pragma unroll
      for (int nt = 0; nt < 2; ++nt)
#pragma unroll
        for (int vv = 0; vv < 8; ++vv)
          pw[(vv + 8 * kh2) * 34 + nt * 16 + n] = f32_to_bf16(st[nt][vv]);
    }
    v16bf pa = load_frag_a(Ps[wave], 34, 0, lane);
    // O += P * V : 8 hd tiles, contraction over 32 keys = one WMMA each.
#pragma unroll
    for (int t = 0; t < 8; ++t) {
      v16bf bfr = load_frag_b(Vs + (t * 16) * 34, 34, 0, lane);
      Oacc[t] = __builtin_amdgcn_wmma_f32_16x16x32_bf16(
          false, pa, false, bfr, (short)0, Oacc[t], false, false);
    }
    __syncthreads();  // before next chunk overwrites Ks/Vs
  }

  // Epilogue: normalize by l, store bf16 to (b, s, h, hd).
  const int n = lane & 15, mo = (lane >> 4) * 8;
#pragma unroll
  for (int t = 0; t < 8; ++t)
#pragma unroll
    for (int vv = 0; vv < 8; ++vv) {
      int row = q0 + vv + mo;
      float val = Oacc[t][vv] / lrow[vv];
      o[((size_t)(b * S_ + row)) * qstride + h * HD_ + t * 16 + n] = f32_to_bf16(val);
    }
}

// ---------------------------------------------------------------------------
// Host-side orchestration.
// ---------------------------------------------------------------------------
extern "C" void kernel_launch(void* const* d_in, const int* in_sizes, int n_in,
                              void* d_out, int out_size, void* d_ws, size_t ws_size,
                              hipStream_t stream) {
  (void)in_sizes; (void)n_in; (void)out_size; (void)ws_size;
  const float* x    = (const float*)d_in[0];
  const float* wq   = (const float*)d_in[1];
  const float* wk   = (const float*)d_in[2];
  const float* wv   = (const float*)d_in[3];
  const float* wo   = (const float*)d_in[4];
  const float* fcos = (const float*)d_in[5];
  const float* fsin = (const float*)d_in[6];
  // d_in[7] = start_pos (0 in this configuration; freqs are absolute positions)

  char* ws = (char*)d_ws;
  size_t off = 0;
  auto alloc16 = [&](size_t elems) {
    unsigned short* p = (unsigned short*)(ws + off);
    off += elems * sizeof(unsigned short);
    return p;
  };
  unsigned short* xb  = alloc16((size_t)MTOT * D_);
  unsigned short* wqb = alloc16((size_t)D_ * D_);
  unsigned short* wkb = alloc16((size_t)D_ * (NKV_ * HD_));
  unsigned short* wvb = alloc16((size_t)D_ * (NKV_ * HD_));
  unsigned short* wob = alloc16((size_t)D_ * D_);
  unsigned short* qb  = alloc16((size_t)MTOT * D_);
  unsigned short* kb  = alloc16((size_t)MTOT * (NKV_ * HD_));
  unsigned short* vb  = alloc16((size_t)MTOT * (NKV_ * HD_));
  unsigned short* ab  = alloc16((size_t)MTOT * D_);

  auto cvt = [&](const float* in, unsigned short* out, size_t nelem) {
    int blocks = (int)((nelem + 2047) / 2048);
    cvt_f32_bf16<<<blocks, 256, 0, stream>>>(in, out, nelem);
  };
  cvt(x,  xb,  (size_t)MTOT * D_);
  cvt(wq, wqb, (size_t)D_ * D_);
  cvt(wk, wkb, (size_t)D_ * (NKV_ * HD_));
  cvt(wv, wvb, (size_t)D_ * (NKV_ * HD_));
  cvt(wo, wob, (size_t)D_ * D_);

  // QKV projections (bf16 out)
  gemm_bf16_wmma<true><<<dim3(D_ / 128, MTOT / 128), 256, 0, stream>>>(
      xb, wqb, qb, MTOT, D_, D_);
  gemm_bf16_wmma<true><<<dim3((NKV_ * HD_) / 128, MTOT / 128), 256, 0, stream>>>(
      xb, wkb, kb, MTOT, NKV_ * HD_, D_);
  gemm_bf16_wmma<true><<<dim3((NKV_ * HD_) / 128, MTOT / 128), 256, 0, stream>>>(
      xb, wvb, vb, MTOT, NKV_ * HD_, D_);

  // RoPE on Q and K
  {
    size_t npq = (size_t)MTOT * NH_ * (HD_ / 2);
    rope_bf16<<<(int)((npq + 255) / 256), 256, 0, stream>>>(qb, fcos, fsin, NH_, npq);
    size_t npk = (size_t)MTOT * NKV_ * (HD_ / 2);
    rope_bf16<<<(int)((npk + 255) / 256), 256, 0, stream>>>(kb, fcos, fsin, NKV_, npk);
  }

  // Flash attention
  attn_wmma<<<dim3(B_ * NH_, S_ / 64), 128, 0, stream>>>(qb, kb, vb, ab);

  // Output projection (fp32 out to d_out)
  gemm_bf16_wmma<false><<<dim3(D_ / 128, MTOT / 128), 256, 0, stream>>>(
      ab, wob, d_out, MTOT, D_, D_);
}
